// MultiHeadRotaryAttention_56573309223931
// MI455X (gfx1250) — matrix operand order repaired and checked by reading, not compiled, not run on hardware
//
#include <hip/hip_runtime.h>

typedef __bf16 bf16_t;
typedef __attribute__((ext_vector_type(16))) __bf16 v16bf;
typedef __attribute__((ext_vector_type(8)))  float  v8f;

#define B_   2
#define S_   2048
#define D_   2048
#define H_   16
#define HD_  128
#define MTOT (B_ * S_)

#define BM 128
#define BN 128
#define BK 32
#define LDT 40    // padded LDS stride for A/B tiles (bf16 elems), 80B rows -> 16B aligned
#define LDC 132   // padded LDS stride for C staging (f32)
#define KT  64    // key tile in attention

__device__ __forceinline__ bf16_t f2bf(float f) { return (bf16_t)f; }

__device__ __forceinline__ v8f wmma_bf16(v16bf a, v16bf b, v8f c) {
  return __builtin_amdgcn_wmma_f32_16x16x32_bf16(false, a, false, b, (short)0, c,
                                                 false, false);
}

// A-matrix 16x32 fragment from row-major [M][K] storage (stride ld). Contiguous runs.
// Lanes 0-15: M=lane, K = {0..7, 16..23}; lanes 16-31: M=lane-16, K = {8..15, 24..31}.
__device__ __forceinline__ v16bf load_a_frag(const bf16_t* p, int ld, int m0, int k0) {
  const int lane = threadIdx.x & 31;
  const int half = lane >> 4;
  const bf16_t* r = p + (size_t)(m0 + (lane & 15)) * ld + k0 + half * 8;
  v16bf a;
#pragma unroll
  for (int i = 0; i < 8; ++i) a[i] = r[i];
#pragma unroll
  for (int i = 0; i < 8; ++i) a[8 + i] = r[16 + i];
  return a;
}

// B-matrix 32x16 fragment when storage is [N][K] (stride ld); two runs of 8.
// Lanes 0-15: N=lane, K = {0..7, 16..23}; lanes 16-31: N=lane-16, K = {8..15, 24..31}.
__device__ __forceinline__ v16bf load_bT_frag(const bf16_t* p, int ld, int k0, int n0) {
  const int lane = threadIdx.x & 31;
  const bf16_t* r = p + (size_t)(n0 + (lane & 15)) * ld + k0 + (lane >> 4) * 8;
  v16bf b;
#pragma unroll
  for (int i = 0; i < 8; ++i) b[i] = r[i];
#pragma unroll
  for (int i = 0; i < 8; ++i) b[8 + i] = r[16 + i];
  return b;
}

__global__ void cvt_f32_bf16(const float* __restrict__ src, bf16_t* __restrict__ dst,
                             int n) {
  int i = blockIdx.x * blockDim.x + threadIdx.x;
  int stride = gridDim.x * blockDim.x;
  for (; i < n; i += stride) dst[i] = f2bf(src[i]);
}

// Transpose-convert a DxD f32 matrix into bf16: dst[n][k] = src[k][n].
__global__ __launch_bounds__(256) void cvt_transpose_bf16(const float* __restrict__ src,
                                                          bf16_t* __restrict__ dst) {
  __shared__ float tile[32][33];
  const int bx = blockIdx.x * 32;  // src col block
  const int by = blockIdx.y * 32;  // src row block
  const int t = threadIdx.x;
  const int r = t >> 3;
  const int c4 = (t & 7) * 4;
#pragma unroll
  for (int j = 0; j < 4; ++j)
    tile[r][c4 + j] = src[(size_t)(by + r) * D_ + bx + c4 + j];
  __syncthreads();
#pragma unroll
  for (int j = 0; j < 4; ++j)
    dst[(size_t)(bx + r) * D_ + by + c4 + j] = f2bf(tile[c4 + j][r]);
}

// ---------------- QKV projection + bias + RoPE --------------------------------------
// X: [M][D] bf16.  Wt: transposed weights [N][K] bf16.
// Q/K out: [b,h,s,hd] bf16 (rotary applied).  V out: transposed [b,h,hd,s] bf16.
__global__ __launch_bounds__(256) void qkv_gemm_rope(
    const bf16_t* __restrict__ X, const bf16_t* __restrict__ Wqt,
    const bf16_t* __restrict__ Wkt, const bf16_t* __restrict__ Wvt,
    const float* __restrict__ bq, const float* __restrict__ bk,
    const float* __restrict__ bv, bf16_t* __restrict__ Qr,
    bf16_t* __restrict__ Kr, bf16_t* __restrict__ Vt) {
  __shared__ bf16_t As[BM * LDT];
  __shared__ bf16_t Bs[BN * LDT];
  __shared__ float  Cs[64 * LDC];

  const int which = blockIdx.z;
  const bf16_t* Wt = (which == 0) ? Wqt : ((which == 1) ? Wkt : Wvt);
  const float* bias = (which == 0) ? bq : ((which == 1) ? bk : bv);

  const int h = blockIdx.x;  // BN == HD: one head per N-tile
  const int n0 = h * BN;
  const int m0 = blockIdx.y * BM;
  const int t = threadIdx.x;
  const int wave = t >> 5, lane = t & 31;
  const int half = lane >> 4, lcol = lane & 15;
  const int waveM = wave & 3;   // 4 slices of 32 rows
  const int waveN = wave >> 2;  // 2 slices of 64 cols

  v8f acc[2][4];
#pragma unroll
  for (int i = 0; i < 2; ++i)
#pragma unroll
    for (int j = 0; j < 4; ++j) acc[i][j] = {};

  const int srow = t >> 1, scol = (t & 1) * 16;  // identical staging map for A and B

  for (int k0 = 0; k0 < D_; k0 += BK) {
    {
      const uint4* g = (const uint4*)(X + (size_t)(m0 + srow) * D_ + k0 + scol);
      uint4* s = (uint4*)(As + srow * LDT + scol);
      s[0] = g[0]; s[1] = g[1];
      const uint4* gw = (const uint4*)(Wt + (size_t)(n0 + srow) * D_ + k0 + scol);
      uint4* sw = (uint4*)(Bs + srow * LDT + scol);
      sw[0] = gw[0]; sw[1] = gw[1];
      if (k0 + BK < D_) {
        __builtin_prefetch(X + (size_t)(m0 + srow) * D_ + k0 + BK + scol, 0, 1);
        __builtin_prefetch(Wt + (size_t)(n0 + srow) * D_ + k0 + BK + scol, 0, 1);
      }
    }
    __syncthreads();
    v16bf af[2];
#pragma unroll
    for (int mt = 0; mt < 2; ++mt) af[mt] = load_a_frag(As, LDT, waveM * 32 + mt * 16, 0);
#pragma unroll
    for (int nt = 0; nt < 4; ++nt) {
      v16bf bfrag = load_bT_frag(Bs, LDT, 0, waveN * 64 + nt * 16);
#pragma unroll
      for (int mt = 0; mt < 2; ++mt)
        acc[mt][nt] = wmma_bf16(af[mt], bfrag, acc[mt][nt]);
    }
    __syncthreads();
  }

  // Epilogue: stage 64 rows at a time through LDS.
#pragma unroll
  for (int rh = 0; rh < 2; ++rh) {
    __syncthreads();
    if ((waveM >> 1) == rh) {
#pragma unroll
      for (int mt = 0; mt < 2; ++mt)
#pragma unroll
        for (int nt = 0; nt < 4; ++nt)
#pragma unroll
          for (int i = 0; i < 8; ++i) {
            int row = (waveM & 1) * 32 + mt * 16 + half * 8 + i;
            Cs[row * LDC + waveN * 64 + nt * 16 + lcol] = acc[mt][nt][i];
          }
    }
    __syncthreads();
    if (which < 2) {
      bf16_t* dst = (which == 0) ? Qr : Kr;
      for (int j = 0; j < 16; ++j) {
        int idx = t + j * 256;  // 64 rows x 64 rotary pairs
        int row = idx >> 6;
        int i = idx & 63;
        int mg = m0 + rh * 64 + row;
        int s = mg & (S_ - 1);
        int bb = mg >> 11;
        float x1 = Cs[row * LDC + i] + bias[n0 + i];
        float x2 = Cs[row * LDC + 64 + i] + bias[n0 + 64 + i];
        // inv_freq = 10000^(-i/64) = exp(-i * ln(10000)/64)
        float fr = (float)s * __expf((float)i * -0.14391156516f);
        float c, sn;
        __sincosf(fr, &sn, &c);
        size_t base = (((size_t)bb * H_ + h) * S_ + s) * HD_;
        dst[base + i] = f2bf(x1 * c - x2 * sn);
        dst[base + 64 + i] = f2bf(x1 * sn + x2 * c);
      }
    } else {
      // V: store transposed [b,h,hd,s]; consecutive lanes -> consecutive s.
      for (int j = 0; j < 32; ++j) {
        int idx = t + j * 256;  // 128 hd x 64 rows
        int hd = idx >> 6;
        int sloc = idx & 63;
        int mg = m0 + rh * 64 + sloc;
        int s = mg & (S_ - 1);
        int bb = mg >> 11;
        float val = Cs[sloc * LDC + hd] + bias[n0 + hd];
        Vt[(((size_t)bb * H_ + h) * HD_ + hd) * S_ + s] = f2bf(val);
      }
    }
  }
}

// ---------------- flash attention: per (b,h), 128-query tile per block -------------
__global__ __launch_bounds__(256) void attn_kernel(const bf16_t* __restrict__ Qr,
                                                   const bf16_t* __restrict__ Kr,
                                                   const bf16_t* __restrict__ Vt,
                                                   bf16_t* __restrict__ Ao) {
  __shared__ bf16_t Ks[KT * HD_];   // [key][hd]
  __shared__ bf16_t Vs[HD_ * KT];   // [hd][key]  (B-operand friendly)
  __shared__ bf16_t Ps[8 * 16 * KT];

  const int bh = blockIdx.y;
  const int b = bh >> 4, h = bh & 15;
  const int q0 = blockIdx.x * 128;
  const int t = threadIdx.x;
  const int wave = t >> 5, lane = t & 31;
  const int half = lane >> 4, lcol = lane & 15;

  const bf16_t* Qh = Qr + (size_t)bh * S_ * HD_;
  const bf16_t* Kh = Kr + (size_t)bh * S_ * HD_;
  const bf16_t* Vh = Vt + (size_t)bh * HD_ * S_;

  // Q fragments (16 rows per wave, 4 K-chunks of 32) in registers for whole pass.
  v16bf qa[4];
  {
    const bf16_t* r = Qh + (size_t)(q0 + wave * 16 + (lane & 15)) * HD_;
#pragma unroll
    for (int kc = 0; kc < 4; ++kc) {
      const bf16_t* rr = r + kc * 32 + half * 8;
#pragma unroll
      for (int i = 0; i < 8; ++i) qa[kc][i] = rr[i];
#pragma unroll
      for (int i = 0; i < 8; ++i) qa[kc][8 + i] = rr[16 + i];
    }
  }

  v8f o[8];
#pragma unroll
  for (int n = 0; n < 8; ++n) o[n] = {};
  float mrow[8], lrow[8];
#pragma unroll
  for (int i = 0; i < 8; ++i) { mrow[i] = -3.0e38f; lrow[i] = 0.0f; }

  const float scale = 0.08838834764831845f;  // 1/sqrt(128)
  bf16_t* Pw = Ps + wave * 16 * KT;

  for (int kt0 = 0; kt0 < S_; kt0 += KT) {
    __syncthreads();
#pragma unroll
    for (int j = 0; j < 4; ++j) {
      int flat = (t + j * 256) * 8;  // 8192 bf16 per tile
      int krow = flat >> 7, kcol = flat & 127;
      *(uint4*)(Ks + flat) = *(const uint4*)(Kh + (size_t)(kt0 + krow) * HD_ + kcol);
      int hd = flat >> 6, key = flat & 63;
      *(uint4*)(Vs + flat) = *(const uint4*)(Vh + (size_t)hd * S_ + kt0 + key);
    }
    __syncthreads();

    // S = Q * K^T (16 x KT per wave); Ks is [key][hd] == [N][K] storage.
    v8f st[4];
#pragma unroll
    for (int n = 0; n < 4; ++n) {
      v8f c = {};
#pragma unroll
      for (int kc = 0; kc < 4; ++kc) {
        v16bf bfrag = load_bT_frag(Ks, HD_, kc * 32, n * 16);
        c = wmma_bf16(qa[kc], bfrag, c);
      }
      st[n] = c;
    }

    float pmax[8];
#pragma unroll
    for (int i = 0; i < 8; ++i) pmax[i] = -3.0e38f;
#pragma unroll
    for (int n = 0; n < 4; ++n)
#pragma unroll
      for (int i = 0; i < 8; ++i) {
        st[n][i] *= scale;
        pmax[i] = fmaxf(pmax[i], st[n][i]);
      }
#pragma unroll
    for (int m = 1; m < 16; m <<= 1)
#pragma unroll
      for (int i = 0; i < 8; ++i) pmax[i] = fmaxf(pmax[i], __shfl_xor(pmax[i], m, 16));

    float psum[8];
#pragma unroll
    for (int i = 0; i < 8; ++i) {
      float mn = fmaxf(mrow[i], pmax[i]);
      float corr = __expf(mrow[i] - mn);
      mrow[i] = mn;
      lrow[i] *= corr;
#pragma unroll
      for (int n = 0; n < 8; ++n) o[n][i] *= corr;
      psum[i] = 0.0f;
    }
#pragma unroll
    for (int n = 0; n < 4; ++n)
#pragma unroll
      for (int i = 0; i < 8; ++i) {
        float p = __expf(st[n][i] - mrow[i]);
        psum[i] += p;
        Pw[(half * 8 + i) * KT + n * 16 + lcol] = f2bf(p);
      }
#pragma unroll
    for (int m = 1; m < 16; m <<= 1)
#pragma unroll
      for (int i = 0; i < 8; ++i) psum[i] += __shfl_xor(psum[i], m, 16);
#pragma unroll
    for (int i = 0; i < 8; ++i) lrow[i] += psum[i];

    // P staging is wave-local: a dscnt wait is sufficient (no block barrier).
    asm volatile("s_wait_dscnt 0x0" ::: "memory");

    v16bf pa[2];
#pragma unroll
    for (int kc = 0; kc < 2; ++kc) pa[kc] = load_a_frag(Pw, KT, 0, kc * 32);
#pragma unroll
    for (int n = 0; n < 8; ++n)
#pragma unroll
      for (int kc = 0; kc < 2; ++kc) {
        v16bf bfrag = load_bT_frag(Vs, KT, kc * 32, n * 16);  // Vs is [hd][key]
        o[n] = wmma_bf16(pa[kc], bfrag, o[n]);
      }
  }

  // Normalize and store attention output in [b, s, h*HD + hd] bf16 (GEMM-ready).
#pragma unroll
  for (int i = 0; i < 8; ++i) {
    int q = q0 + wave * 16 + half * 8 + i;
    size_t base = ((size_t)b * S_ + q) * D_ + h * HD_;
    float inv = 1.0f / lrow[i];
#pragma unroll
    for (int n = 0; n < 8; ++n) Ao[base + n * 16 + lcol] = f2bf(o[n][i] * inv);
  }
}

// ---------------- output projection: attn(bf16) x Wo^T(bf16) + bo -> f32 -----------
__global__ __launch_bounds__(256) void out_gemm(const bf16_t* __restrict__ Abf,
                                                const bf16_t* __restrict__ Wot,
                                                const float* __restrict__ bo,
                                                float* __restrict__ out) {
  __shared__ bf16_t As[BM * LDT];
  __shared__ bf16_t Bs[BN * LDT];

  const int n0 = blockIdx.x * BN;
  const int m0 = blockIdx.y * BM;
  const int t = threadIdx.x;
  const int wave = t >> 5, lane = t & 31;
  const int half = lane >> 4, lcol = lane & 15;
  const int waveM = wave & 3;
  const int waveN = wave >> 2;

  v8f acc[2][4];
#pragma unroll
  for (int i = 0; i < 2; ++i)
#pragma unroll
    for (int j = 0; j < 4; ++j) acc[i][j] = {};

  const int srow = t >> 1, scol = (t & 1) * 16;

  for (int k0 = 0; k0 < D_; k0 += BK) {
    {
      const uint4* g = (const uint4*)(Abf + (size_t)(m0 + srow) * D_ + k0 + scol);
      uint4* s = (uint4*)(As + srow * LDT + scol);
      s[0] = g[0]; s[1] = g[1];
      const uint4* gw = (const uint4*)(Wot + (size_t)(n0 + srow) * D_ + k0 + scol);
      uint4* sw = (uint4*)(Bs + srow * LDT + scol);
      sw[0] = gw[0]; sw[1] = gw[1];
      if (k0 + BK < D_) {
        __builtin_prefetch(Abf + (size_t)(m0 + srow) * D_ + k0 + BK + scol, 0, 1);
        __builtin_prefetch(Wot + (size_t)(n0 + srow) * D_ + k0 + BK + scol, 0, 1);
      }
    }
    __syncthreads();
    v16bf af[2];
#pragma unroll
    for (int mt = 0; mt < 2; ++mt) af[mt] = load_a_frag(As, LDT, waveM * 32 + mt * 16, 0);
#pragma unroll
    for (int nt = 0; nt < 4; ++nt) {
      v16bf bfrag = load_bT_frag(Bs, LDT, 0, waveN * 64 + nt * 16);
#pragma unroll
      for (int mt = 0; mt < 2; ++mt)
        acc[mt][nt] = wmma_bf16(af[mt], bfrag, acc[mt][nt]);
    }
    __syncthreads();
  }

#pragma unroll
  for (int mt = 0; mt < 2; ++mt)
#pragma unroll
    for (int nt = 0; nt < 4; ++nt)
#pragma unroll
      for (int i = 0; i < 8; ++i) {
        int m = m0 + waveM * 32 + mt * 16 + half * 8 + i;
        int n = n0 + waveN * 64 + nt * 16 + lcol;
        out[(size_t)m * D_ + n] = acc[mt][nt][i] + bo[n];
      }
}

extern "C" void kernel_launch(void* const* d_in, const int* in_sizes, int n_in,
                              void* d_out, int out_size, void* d_ws, size_t ws_size,
                              hipStream_t stream) {
  (void)in_sizes; (void)n_in; (void)out_size; (void)ws_size;
  const float* x  = (const float*)d_in[0];
  const float* Wq = (const float*)d_in[1];
  const float* bq = (const float*)d_in[2];
  const float* Wk = (const float*)d_in[3];
  const float* bk = (const float*)d_in[4];
  const float* Wv = (const float*)d_in[5];
  const float* bv = (const float*)d_in[6];
  const float* Wo = (const float*)d_in[7];
  const float* bo = (const float*)d_in[8];
  float* out = (float*)d_out;

  const size_t MD = (size_t)MTOT * D_;
  const size_t DD = (size_t)D_ * D_;
  bf16_t* p = (bf16_t*)d_ws;
  bf16_t* xbf = p; p += MD;
  bf16_t* Wqb = p; p += DD;   // transposed
  bf16_t* Wkb = p; p += DD;   // transposed
  bf16_t* Wvb = p; p += DD;   // transposed
  bf16_t* Wob = p; p += DD;   // transposed
  bf16_t* Qr  = p; p += MD;   // [b,h,s,hd]
  bf16_t* Kr  = p; p += MD;   // [b,h,s,hd]
  bf16_t* Vt  = p; p += MD;   // [b,h,hd,s]
  bf16_t* Ao  = p; p += MD;   // [b,s,d]

  cvt_f32_bf16<<<2048, 256, 0, stream>>>(x, xbf, (int)MD);
  dim3 tg(D_ / 32, D_ / 32);
  cvt_transpose_bf16<<<tg, 256, 0, stream>>>(Wq, Wqb);
  cvt_transpose_bf16<<<tg, 256, 0, stream>>>(Wk, Wkb);
  cvt_transpose_bf16<<<tg, 256, 0, stream>>>(Wv, Wvb);
  cvt_transpose_bf16<<<tg, 256, 0, stream>>>(Wo, Wob);

  qkv_gemm_rope<<<dim3(D_ / BN, MTOT / BM, 3), 256, 0, stream>>>(
      xbf, Wqb, Wkb, Wvb, bq, bk, bv, Qr, Kr, Vt);

  attn_kernel<<<dim3(S_ / 128, B_ * H_), 256, 0, stream>>>(Qr, Kr, Vt, Ao);

  out_gemm<<<dim3(D_ / BN, MTOT / BM), 256, 0, stream>>>(Ao, Wob, bo, out);
}
